// Ligand_SE3_20770461843836
// MI455X (gfx1250) — compile-verified, matrix operand order checked by reading, not run on hardware
//
#include <hip/hip_runtime.h>
#include <hip/hip_bf16.h>
#include <math.h>

// ---------------------------------------------------------------------------
// SE(3)-equivariant GNN forward for MI455X (gfx1250, wave32, WMMA).
// Dead code eliminated vs reference:
//   - layer0: k1/k2 never used by layer0 logits  -> not computed
//   - layer1: v1/v2/f1n/f2n never reach output   -> not computed
// All dense 32-wide GEMMs run as V_WMMA_F32_16X16X32_F16 (f16 in, f32 acc).
// Round 2: fragment-order B staging (v16h LDS loads), contiguous-k A
// assembly (v8h/float4), transposed C scratch (v8f LDS stores), coalesced
// float4 value writes, minimal barriers.
// ---------------------------------------------------------------------------

#define NN_NODES 10000
#define NN_EDGES 160000
#define L0IN     15

typedef __attribute__((ext_vector_type(16))) _Float16 v16h;
typedef __attribute__((ext_vector_type(8)))  _Float16 v8h;
typedef __attribute__((ext_vector_type(8)))  float    v8f;

__device__ __forceinline__ v8f wmma16x16x32(v16h a, v16h b, v8f c) {
  // 8 args: (neg_a, A, neg_b, B, c_mod, C, reuse_a, reuse_b)
  return __builtin_amdgcn_wmma_f32_16x16x32_f16(false, a, false, b, (short)0, c,
                                                false, false);
}

// K index for 16-bit 16x32 A/B fragment element j in this lane (ISA 7.12.2).
// Contiguous in j: k = base + j (j<8), k = 16 + base + (j-8) (j>=8).
__device__ __forceinline__ int frag_k(int lane, int j) {
  int base = (lane >= 16) ? 8 : 0;
  return (j < 8) ? (base + j) : (16 + base + (j - 8));
}

__device__ __forceinline__ void atomicMaxF(float* addr, float v) {
  if (v >= 0.f) atomicMax((int*)addr, __float_as_int(v));
  else          atomicMin((unsigned int*)addr, __float_as_uint(v));
}

// ---------------------------------------------------------------------------
// Generic strided GEMM: C = act(A@B [+ A2@B2] [+ bias]) with K,K2,N <= 32.
// B staged in LDS in fragment order -> one v16h load per WMMA operand.
// Block = 128 threads = 4 waves, each wave one 16-row M-tile.
// ---------------------------------------------------------------------------
__global__ void k_gemm(const float* __restrict__ A, int rsA, int csA, int offA, int K1,
                       const float* __restrict__ B,
                       const float* __restrict__ A2, int rsA2, int csA2, int offA2, int K2,
                       const float* __restrict__ B2,
                       const float* __restrict__ bias,
                       float* __restrict__ Co, int rsC, int csC, int offC,
                       int M, int N, int act)
{
  __shared__ v16h lBf[2][2][32];  // [mat][ntile][lane] fragment-order
  _Float16* pB = (_Float16*)lBf;
  int t = threadIdx.x;
  for (int idx = t; idx < 2 * 2 * 32 * 16; idx += 128) {
    int j = idx & 15, l = (idx >> 4) & 31, tile = (idx >> 9) & 1, mat = idx >> 10;
    int k = frag_k(l, j);
    int n = tile * 16 + (l & 15);
    const float* Bp = mat ? B2 : B;
    int KK = mat ? K2 : K1;
    float v = (Bp != nullptr && k < KK && n < N) ? Bp[k * N + n] : 0.f;
    pB[idx] = (_Float16)v;
  }
  __syncthreads();

  int wave = t >> 5, lane = t & 31;
  int row0 = (blockIdx.x * 4 + wave) * 16;
  int r = row0 + (lane & 15);
  bool rok = (r < M);
  int rc = rok ? r : 0;
  int base = (lane < 16) ? 0 : 8;

  auto loadA = [&](const float* Ap, int rs, int cs, int of, int KK) {
    v16h af;
    if (rok && cs == 1 && KK == 32) {  // aligned 32-float rows: float4 path
      const float* p = Ap + of + (size_t)rc * rs;
      float4 fa = *(const float4*)(p + base);
      float4 fb = *(const float4*)(p + base + 4);
      float4 fc = *(const float4*)(p + 16 + base);
      float4 fd = *(const float4*)(p + 16 + base + 4);
      float tmp[16] = {fa.x, fa.y, fa.z, fa.w, fb.x, fb.y, fb.z, fb.w,
                       fc.x, fc.y, fc.z, fc.w, fd.x, fd.y, fd.z, fd.w};
#pragma unroll
      for (int j = 0; j < 16; ++j) af[j] = (_Float16)tmp[j];
    } else {
#pragma unroll
      for (int j = 0; j < 16; ++j) {
        int k = frag_k(lane, j);
        af[j] = (_Float16)((rok && k < KK) ? Ap[of + (size_t)rc * rs + k * cs] : 0.f);
      }
    }
    return af;
  };
  v16h af = loadA(A, rsA, csA, offA, K1);
  v16h a2f;
  if (A2 != nullptr) a2f = loadA(A2, rsA2, csA2, offA2, K2);

  int ntiles = (N + 15) >> 4;
  for (int nt = 0; nt < ntiles; ++nt) {
    int ncol = nt * 16 + (lane & 15);
    v8f acc = {};
    acc = wmma16x16x32(af, lBf[0][nt][lane], acc);
    if (A2 != nullptr) acc = wmma16x16x32(a2f, lBf[1][nt][lane], acc);
#pragma unroll
    for (int rr = 0; rr < 8; ++rr) {
      int row = row0 + base + rr;
      if (row < M && ncol < N) {
        float v = acc[rr];
        if (bias) v += bias[ncol];
        if (act == 1) v = fmaxf(v, 0.f);
        Co[offC + (size_t)row * rsC + ncol * csC] = v;
      }
    }
  }
}

// ---------------------------------------------------------------------------
// Edge geometry: rel, r, Y1 (l=1), Y2 (l=2).
// ---------------------------------------------------------------------------
__global__ void k_geom(const float* __restrict__ pos, const int* __restrict__ src,
                       const int* __restrict__ dst,
                       float* __restrict__ Y1, float* __restrict__ Y2,
                       float* __restrict__ rr, int E)
{
  int e = blockIdx.x * blockDim.x + threadIdx.x;
  if (e >= E) return;
  int s = src[e], d = dst[e];
  float rx = pos[d * 3 + 0] - pos[s * 3 + 0];
  float ry = pos[d * 3 + 1] - pos[s * 3 + 1];
  float rz = pos[d * 3 + 2] - pos[s * 3 + 2];
  float r = sqrtf(rx * rx + ry * ry + rz * rz);
  float inv = 1.f / (r + 1e-6f);
  float ux = rx * inv, uy = ry * inv, uz = rz * inv;
  const float s3 = 1.7320508075688772f;
  Y1[e * 3 + 0] = ux; Y1[e * 3 + 1] = uy; Y1[e * 3 + 2] = uz;
  Y2[e * 5 + 0] = s3 * ux * uy;
  Y2[e * 5 + 1] = s3 * uy * uz;
  Y2[e * 5 + 2] = 0.5f * (3.f * uz * uz - 1.f);
  Y2[e * 5 + 3] = s3 * ux * uz;
  Y2[e * 5 + 4] = 0.5f * s3 * (ux * ux - uy * uy);
  rr[e] = r;
}

__global__ void k_fill(float* __restrict__ p, float v, int n)
{
  int i = blockIdx.x * blockDim.x + threadIdx.x;
  if (i < n) p[i] = v;
}

// ---------------------------------------------------------------------------
// Layer-0 edge kernel. Wave = 16-edge tile. Inline radial MLP (6->16->90),
// WMMA mixes for live slices {k0, v0, v1c, v2c}, logits = q0 . k0 / sqrt(8).
// v1/v2 stored as [e][m][c] so value writes are contiguous float4s.
// ---------------------------------------------------------------------------
__global__ void k_l0_edge(const float* __restrict__ f0,    // node_attr N x 15
                          const float* __restrict__ eattr, // E x 5
                          const int* __restrict__ src, const int* __restrict__ dst,
                          const float* __restrict__ q0,    // N x 32
                          const float* __restrict__ Y1, const float* __restrict__ Y2,
                          const float* __restrict__ rr,
                          const float* __restrict__ Wr1, const float* __restrict__ br1,
                          const float* __restrict__ Wr2, const float* __restrict__ br2,
                          const float* __restrict__ Wk0, const float* __restrict__ Wv0,
                          const float* __restrict__ Wv1, const float* __restrict__ Wv2,
                          float* __restrict__ v0, float* __restrict__ v1,
                          float* __restrict__ v2, float* __restrict__ lg, int E)
{
  __shared__ v16h    lBf[4][2][32];                              // 8 KB
  __shared__ _Float16 lw[4][16][6][16];                          // 12 KB
  __shared__ float   lkT[4][32][16] __attribute__((aligned(32))); // k0^T, 8 KB
  __shared__ float   lsv[4][32][16] __attribute__((aligned(32))); // v stage, 8 KB
  __shared__ int     lsrc[4][16], ldst[4][16];
  __shared__ float   lY1s[4][16][3], lY2s[4][16][5];

  int t = threadIdx.x, wave = t >> 5, lane = t & 31;
  // --- B staging in fragment order (K padded 15 -> 32) ---
  {
    const float* Ws[4] = {Wk0, Wv0, Wv1, Wv2};
    _Float16* pB = (_Float16*)lBf;
    for (int idx = t; idx < 4 * 2 * 32 * 16; idx += 128) {
      int j = idx & 15, l = (idx >> 4) & 31, tile = (idx >> 9) & 1, mat = idx >> 10;
      int k = frag_k(l, j);
      int n = tile * 16 + (l & 15);
      pB[idx] = (k < L0IN) ? (_Float16)Ws[mat][k * 32 + n] : (_Float16)0.f;
    }
  }
  int e0 = (blockIdx.x * 4 + wave) * 16;
  // --- per-edge setup + radial MLP (wave-private LDS) ---
  {
    int i = lane & 15;
    int e = e0 + i; if (e >= E) e = E - 1;
    if (lane < 16) {
      lsrc[wave][i] = src[e]; ldst[wave][i] = dst[e];
      for (int m = 0; m < 3; ++m) lY1s[wave][i][m] = Y1[e * 3 + m];
      for (int m = 0; m < 5; ++m) lY2s[wave][i][m] = Y2[e * 5 + m];
      for (int sl = 0; sl < 6; ++sl) lw[wave][i][sl][15] = (_Float16)0.f;
    }
    float rf[6];
    rf[0] = rr[e];
    for (int f = 0; f < 5; ++f) rf[1 + f] = eattr[e * 5 + f];
    float h[16];
    for (int j = 0; j < 16; ++j) {
      float a = br1[j];
      for (int f = 0; f < 6; ++f) a += rf[f] * Wr1[f * 16 + j];
      h[j] = fmaxf(a, 0.f);
    }
    int half = lane >> 4;
    for (int jj = 0; jj < 45; ++jj) {
      int jidx = half * 45 + jj;
      int sl = jidx / 15, k = jidx % 15;
      float a = br2[jidx];
      for (int m = 0; m < 16; ++m) a += h[m] * Wr2[m * 90 + jidx];
      lw[wave][i][sl][k] = (_Float16)a;
    }
  }
  __syncthreads();  // lBf block-wide; everything after is wave-private

  int rloc = lane & 15;
  int nl   = lane & 15;
  int srcn = lsrc[wave][rloc];
  int base = (lane < 16) ? 0 : 8;
  const int slices[4] = {0, 1, 3, 5};  // k0, v0, v1-coef, v2-coef

  for (int s = 0; s < 4; ++s) {
    int sl = slices[s];
    // A fragment: contiguous k; only the low-8 chunk can be non-zero (K=15).
    v16h af;
    {
      v8h wv = *(const v8h*)&lw[wave][rloc][sl][base];
#pragma unroll
      for (int j = 0; j < 8; ++j) {
        int k = base + j;
        float fv = (k < L0IN) ? f0[(size_t)srcn * L0IN + k] : 0.f;
        af[j] = (_Float16)(fv * (float)wv[j]);
      }
#pragma unroll
      for (int j = 8; j < 16; ++j) af[j] = (_Float16)0.f;  // k in [16+base, 24+base) >= 15
    }
    for (int nt = 0; nt < 2; ++nt) {
      v8f acc = {};
      acc = wmma16x16x32(af, lBf[s][nt][lane], acc);
      int ncol = nt * 16 + nl;
      if (s == 0) *(v8f*)&lkT[wave][ncol][base] = acc;  // 8 contiguous rows
      else        *(v8f*)&lsv[wave][ncol][base] = acc;
    }
    // transposed, coalesced value writes (wave-private; DS in-order per wave)
    if (s > 0) {
      int i = lane & 15, half = lane >> 4;
      int e = e0 + i;
      if (e < E) {
        if (s == 1) {
#pragma unroll
          for (int qq = 0; qq < 4; ++qq) {
            int c0 = half * 16 + qq * 4;
            float4 o = {lsv[wave][c0 + 0][i], lsv[wave][c0 + 1][i],
                        lsv[wave][c0 + 2][i], lsv[wave][c0 + 3][i]};
            *(float4*)&v0[(size_t)e * 32 + c0] = o;
          }
        } else if (s == 2) {
          for (int m = 0; m < 3; ++m) {
            float ym = lY1s[wave][i][m];
#pragma unroll
            for (int qq = 0; qq < 4; ++qq) {
              int c0 = half * 16 + qq * 4;
              float4 o = {lsv[wave][c0 + 0][i] * ym, lsv[wave][c0 + 1][i] * ym,
                          lsv[wave][c0 + 2][i] * ym, lsv[wave][c0 + 3][i] * ym};
              *(float4*)&v1[(size_t)e * 96 + m * 32 + c0] = o;
            }
          }
        } else {
          for (int m = 0; m < 5; ++m) {
            float ym = lY2s[wave][i][m];
#pragma unroll
            for (int qq = 0; qq < 4; ++qq) {
              int c0 = half * 16 + qq * 4;
              float4 o = {lsv[wave][c0 + 0][i] * ym, lsv[wave][c0 + 1][i] * ym,
                          lsv[wave][c0 + 2][i] * ym, lsv[wave][c0 + 3][i] * ym};
              *(float4*)&v2[(size_t)e * 160 + m * 32 + c0] = o;
            }
          }
        }
      }
    }
  }
  // logits = q0 . k0 / sqrt(8)
  for (int task = lane; task < 64; task += 32) {
    int i = task & 15, hh = task >> 4;
    int e = e0 + i;
    if (e < E) {
      int d = ldst[wave][i];
      float sacc = 0.f;
#pragma unroll
      for (int c = 0; c < 8; ++c)
        sacc += q0[(size_t)d * 32 + hh * 8 + c] * lkT[wave][hh * 8 + c][i];
      lg[e * 4 + hh] = sacc * 0.35355339059f;
    }
  }
}

// ---------------------------------------------------------------------------
// Layer-1 edge kernel. Wave = 16-edge tile. Inline radial MLP (6->16->320,
// live slices {0,1,2,3,6,7}). WMMA for k0,v0,k1y,k2y and per-component
// k1s (x3), k2s (x5). logits = (q0.k0 + q1.k1 + q2.k2)/sqrt(72). Only v0 kept.
// ---------------------------------------------------------------------------
__global__ void k_l1_edge(const float* __restrict__ f0A, const float* __restrict__ f1A,
                          const float* __restrict__ f2A,
                          const float* __restrict__ eattr,
                          const int* __restrict__ src, const int* __restrict__ dst,
                          const float* __restrict__ q0, const float* __restrict__ q1,
                          const float* __restrict__ q2,
                          const float* __restrict__ Y1, const float* __restrict__ Y2,
                          const float* __restrict__ rr,
                          const float* __restrict__ Wr1, const float* __restrict__ br1,
                          const float* __restrict__ Wr2, const float* __restrict__ br2,
                          const float* __restrict__ Wk0, const float* __restrict__ Wv0,
                          const float* __restrict__ Wk1s, const float* __restrict__ Wk1y,
                          const float* __restrict__ Wk2s, const float* __restrict__ Wk2y,
                          float* __restrict__ v0, float* __restrict__ lg, int E)
{
  __shared__ v16h    lBf[6][2][32];                               // 12 KB
  __shared__ _Float16 lw[4][16][6][32];                           // 24 KB
  __shared__ float   lkT[4][32][16]  __attribute__((aligned(32))); // 8 KB
  __shared__ float   lkyT[4][32][16] __attribute__((aligned(32))); // 8 KB
  __shared__ float   llog[4][16][4];
  __shared__ int     lsrc[4][16], ldst[4][16];
  __shared__ float   lY1s[4][16][3], lY2s[4][16][5];

  int t = threadIdx.x, wave = t >> 5, lane = t & 31;
  {
    const float* Ws[6] = {Wk0, Wv0, Wk1s, Wk1y, Wk2s, Wk2y};
    _Float16* pB = (_Float16*)lBf;
    for (int idx = t; idx < 6 * 2 * 32 * 16; idx += 128) {
      int j = idx & 15, l = (idx >> 4) & 31, tile = (idx >> 9) & 1, mat = idx >> 10;
      int k = frag_k(l, j);
      int n = tile * 16 + (l & 15);
      pB[idx] = (_Float16)Ws[mat][k * 32 + n];
    }
  }
  int e0 = (blockIdx.x * 4 + wave) * 16;
  {
    int i = lane & 15;
    int e = e0 + i; if (e >= E) e = E - 1;
    if (lane < 16) {
      lsrc[wave][i] = src[e]; ldst[wave][i] = dst[e];
      for (int m = 0; m < 3; ++m) lY1s[wave][i][m] = Y1[e * 3 + m];
      for (int m = 0; m < 5; ++m) lY2s[wave][i][m] = Y2[e * 5 + m];
      for (int hh = 0; hh < 4; ++hh) llog[wave][i][hh] = 0.f;
    }
    float rf[6];
    rf[0] = rr[e];
    for (int f = 0; f < 5; ++f) rf[1 + f] = eattr[e * 5 + f];
    float h[16];
    for (int j = 0; j < 16; ++j) {
      float a = br1[j];
      for (int f = 0; f < 6; ++f) a += rf[f] * Wr1[f * 16 + j];
      h[j] = fmaxf(a, 0.f);
    }
    const int imap[6] = {0, 1, 2, 3, 6, 7};  // live w-slices
    int half = lane >> 4;
    for (int jj = 0; jj < 96; ++jj) {
      int jidx = half * 96 + jj;
      int slot = jidx >> 5, k = jidx & 31;
      int col = imap[slot] * 32 + k;
      float a = br2[col];
      for (int m = 0; m < 16; ++m) a += h[m] * Wr2[m * 320 + col];
      lw[wave][i][slot][k] = (_Float16)a;
    }
  }
  __syncthreads();  // lBf block-wide; rest is wave-private (DS in-order)

  int rloc = lane & 15;
  int nl   = lane & 15;
  int srcn = lsrc[wave][rloc];
  int base = (lane < 16) ? 0 : 8;
  const float invsc = 0.11785113020f;  // 1/sqrt(72)

  // register cache of Y for this lane's 8 accumulator rows (base..base+7)
  float y1c[3][8], y2c[5][8];
#pragma unroll
  for (int m = 0; m < 3; ++m)
#pragma unroll
    for (int rw = 0; rw < 8; ++rw) y1c[m][rw] = lY1s[wave][base + rw][m];
#pragma unroll
  for (int m = 0; m < 5; ++m)
#pragma unroll
    for (int rw = 0; rw < 8; ++rw) y2c[m][rw] = lY2s[wave][base + rw][m];

  // A fragment from f0A row (128B aligned) * radial slice (contiguous k)
  auto afrag0 = [&](int slot) {
    v8h wa = *(const v8h*)&lw[wave][rloc][slot][base];
    v8h wb = *(const v8h*)&lw[wave][rloc][slot][16 + base];
    const float* p = f0A + (size_t)srcn * 32;
    float4 fa = *(const float4*)(p + base);
    float4 fb = *(const float4*)(p + base + 4);
    float4 fc = *(const float4*)(p + 16 + base);
    float4 fd = *(const float4*)(p + 16 + base + 4);
    float tmp[16] = {fa.x, fa.y, fa.z, fa.w, fb.x, fb.y, fb.z, fb.w,
                     fc.x, fc.y, fc.z, fc.w, fd.x, fd.y, fd.z, fd.w};
    v16h af;
#pragma unroll
    for (int j = 0; j < 8; ++j)  af[j] = (_Float16)(tmp[j] * (float)wa[j]);
#pragma unroll
    for (int j = 8; j < 16; ++j) af[j] = (_Float16)(tmp[j] * (float)wb[j - 8]);
    return af;
  };

  // ---- k0 -> lkT; llog += q0.k0 ----
  {
    v16h af = afrag0(0);
    for (int nt = 0; nt < 2; ++nt) {
      v8f acc = {};
      acc = wmma16x16x32(af, lBf[0][nt][lane], acc);
      *(v8f*)&lkT[wave][nt * 16 + nl][base] = acc;
    }
    for (int task = lane; task < 64; task += 32) {
      int i = task & 15, hh = task >> 4;
      int d = ldst[wave][i];
      float sacc = 0.f;
#pragma unroll
      for (int c = 0; c < 8; ++c)
        sacc += q0[(size_t)d * 32 + hh * 8 + c] * lkT[wave][hh * 8 + c][i];
      llog[wave][i][hh] += sacc;
    }
  }

  // ---- v0: stage through lkT (k0 consumed), coalesced float4 writes ----
  {
    v16h af = afrag0(1);
    for (int nt = 0; nt < 2; ++nt) {
      v8f acc = {};
      acc = wmma16x16x32(af, lBf[1][nt][lane], acc);
      *(v8f*)&lkT[wave][nt * 16 + nl][base] = acc;
    }
    int i = lane & 15, half = lane >> 4;
    int e = e0 + i;
    if (e < E) {
#pragma unroll
      for (int qq = 0; qq < 4; ++qq) {
        int c0 = half * 16 + qq * 4;
        float4 o = {lkT[wave][c0 + 0][i], lkT[wave][c0 + 1][i],
                    lkT[wave][c0 + 2][i], lkT[wave][c0 + 3][i]};
        *(float4*)&v0[(size_t)e * 32 + c0] = o;
      }
    }
  }

  // ---- k1y coefficient -> lkyT ----
  {
    v16h af = afrag0(3);
    for (int nt = 0; nt < 2; ++nt) {
      v8f acc = {};
      acc = wmma16x16x32(af, lBf[3][nt][lane], acc);
      *(v8f*)&lkyT[wave][nt * 16 + nl][base] = acc;
    }
  }
  // ---- k1 per component m: (f1s_m * w2)@Wk1s + k1y*Y1[m]; llog += q1.k1 ----
  {
    v8h wa = *(const v8h*)&lw[wave][rloc][2][base];
    v8h wb = *(const v8h*)&lw[wave][rloc][2][16 + base];
    for (int m = 0; m < 3; ++m) {
      v16h af;
#pragma unroll
      for (int j = 0; j < 8; ++j) {
        int k = base + j;
        af[j] = (_Float16)(f1A[(size_t)srcn * 96 + k * 3 + m] * (float)wa[j]);
      }
#pragma unroll
      for (int j = 8; j < 16; ++j) {
        int k = 16 + base + (j - 8);
        af[j] = (_Float16)(f1A[(size_t)srcn * 96 + k * 3 + m] * (float)wb[j - 8]);
      }
      for (int nt = 0; nt < 2; ++nt) {
        v8f acc = {};
        acc = wmma16x16x32(af, lBf[2][nt][lane], acc);
        v8f ky = *(const v8f*)&lkyT[wave][nt * 16 + nl][base];
        v8f ov;
#pragma unroll
        for (int rw = 0; rw < 8; ++rw) ov[rw] = acc[rw] + ky[rw] * y1c[m][rw];
        *(v8f*)&lkT[wave][nt * 16 + nl][base] = ov;
      }
      for (int task = lane; task < 64; task += 32) {
        int i = task & 15, hh = task >> 4;
        int d = ldst[wave][i];
        float sacc = 0.f;
#pragma unroll
        for (int c = 0; c < 8; ++c)
          sacc += q1[(size_t)d * 96 + (hh * 8 + c) * 3 + m] * lkT[wave][hh * 8 + c][i];
        llog[wave][i][hh] += sacc;
      }
    }
  }

  // ---- k2y coefficient -> lkyT ----
  {
    v16h af = afrag0(5);
    for (int nt = 0; nt < 2; ++nt) {
      v8f acc = {};
      acc = wmma16x16x32(af, lBf[5][nt][lane], acc);
      *(v8f*)&lkyT[wave][nt * 16 + nl][base] = acc;
    }
  }
  // ---- k2 per component m (5); llog += q2.k2 ----
  {
    v8h wa = *(const v8h*)&lw[wave][rloc][4][base];
    v8h wb = *(const v8h*)&lw[wave][rloc][4][16 + base];
    for (int m = 0; m < 5; ++m) {
      v16h af;
#pragma unroll
      for (int j = 0; j < 8; ++j) {
        int k = base + j;
        af[j] = (_Float16)(f2A[(size_t)srcn * 160 + k * 5 + m] * (float)wa[j]);
      }
#pragma unroll
      for (int j = 8; j < 16; ++j) {
        int k = 16 + base + (j - 8);
        af[j] = (_Float16)(f2A[(size_t)srcn * 160 + k * 5 + m] * (float)wb[j - 8]);
      }
      for (int nt = 0; nt < 2; ++nt) {
        v8f acc = {};
        acc = wmma16x16x32(af, lBf[4][nt][lane], acc);
        v8f ky = *(const v8f*)&lkyT[wave][nt * 16 + nl][base];
        v8f ov;
#pragma unroll
        for (int rw = 0; rw < 8; ++rw) ov[rw] = acc[rw] + ky[rw] * y2c[m][rw];
        *(v8f*)&lkT[wave][nt * 16 + nl][base] = ov;
      }
      for (int task = lane; task < 64; task += 32) {
        int i = task & 15, hh = task >> 4;
        int d = ldst[wave][i];
        float sacc = 0.f;
#pragma unroll
        for (int c = 0; c < 8; ++c)
          sacc += q2[(size_t)d * 160 + (hh * 8 + c) * 5 + m] * lkT[wave][hh * 8 + c][i];
        llog[wave][i][hh] += sacc;
      }
    }
  }

  for (int task = lane; task < 64; task += 32) {
    int i = task & 15, hh = task >> 4;
    int e = e0 + i;
    if (e < E) lg[e * 4 + hh] = llog[wave][i][hh] * invsc;
  }
}

// ---------------------------------------------------------------------------
// Segment softmax (dst only sorted, no CSR): atomic max, then exp + atomic sum.
// ---------------------------------------------------------------------------
__global__ void k_smax_max(const float* __restrict__ lg, const int* __restrict__ dst,
                           float* __restrict__ mx, int E)
{
  int idx = blockIdx.x * blockDim.x + threadIdx.x;
  if (idx >= E * 4) return;
  int e = idx >> 2, hh = idx & 3;
  atomicMaxF(&mx[dst[e] * 4 + hh], lg[idx]);
}

__global__ void k_smax_exp(const float* __restrict__ lg, const int* __restrict__ dst,
                           const float* __restrict__ mx, float* __restrict__ aa,
                           float* __restrict__ zz, int E)
{
  int idx = blockIdx.x * blockDim.x + threadIdx.x;
  if (idx >= E * 4) return;
  int e = idx >> 2, hh = idx & 3;
  float ex = __expf(lg[idx] - mx[dst[e] * 4 + hh]);
  aa[idx] = ex;
  atomicAdd(&zz[dst[e] * 4 + hh], ex);
}

// Scatter-aggregate a-weighted values into destination nodes.
// v1/v2 use [e][m][c] layout (contiguous c).
__global__ void k_agg(const float* __restrict__ aa, const float* __restrict__ zz,
                      const int* __restrict__ dst,
                      const float* __restrict__ v0, const float* __restrict__ v1,
                      const float* __restrict__ v2,
                      float* __restrict__ o0, float* __restrict__ o1,
                      float* __restrict__ o2, int E, int full)
{
  int idx = blockIdx.x * blockDim.x + threadIdx.x;
  if (idx >= E * 32) return;
  int e = idx >> 5, c = idx & 31, hh = c >> 3;
  int d = dst[e];
  float a = aa[e * 4 + hh] / (zz[d * 4 + hh] + 1e-9f);
  atomicAdd(&o0[d * 32 + c], v0[(size_t)e * 32 + c] * a);
  if (full) {
    __builtin_prefetch(&v2[(size_t)e * 160 + c], 0, 1);
    for (int m = 0; m < 3; ++m)
      atomicAdd(&o1[d * 96 + c * 3 + m], v1[(size_t)e * 96 + m * 32 + c] * a);
    for (int m = 0; m < 5; ++m)
      atomicAdd(&o2[d * 160 + c * 5 + m], v2[(size_t)e * 160 + m * 32 + c] * a);
  }
}

// Equivariant gate: x[n,c,:] *= sigmoid(||x[n,c,:]|| * a[c] + b[c])
__global__ void k_gate(float* __restrict__ x, const float* __restrict__ a,
                       const float* __restrict__ b, int Nn, int M)
{
  int idx = blockIdx.x * blockDim.x + threadIdx.x;
  if (idx >= Nn * 32) return;
  int n = idx >> 5, c = idx & 31;
  float ss = 1e-9f;
  for (int m = 0; m < M; ++m) {
    float v = x[((size_t)n * 32 + c) * M + m];
    ss += v * v;
  }
  float nrm = sqrtf(ss);
  float g = 1.f / (1.f + __expf(-(nrm * a[c] + b[c])));
  for (int m = 0; m < M; ++m) x[((size_t)n * 32 + c) * M + m] *= g;
}

// ---------------------------------------------------------------------------
// Host-side orchestration.
// Input layout: setup_inputs() order, params flattened as jax pytree
// (dict keys sorted lexicographically; 'W' < 'a' in ASCII).
// ---------------------------------------------------------------------------
enum ParamIdx {
  IN_NODE_ATTR = 0, IN_EDGE_ATTR, IN_POS, IN_SRC, IN_DST,
  P_WOUT, P_BOUT,
  L0_WK0, L0_WK1, L0_WK2, L0_WO0, L0_WO1, L0_WO2, L0_WQ0, L0_WR1, L0_WR2,
  L0_WS0, L0_WV0, L0_WV1, L0_WV2, L0_A1, L0_A2, L0_B1, L0_B2, L0_BR1, L0_BR2,
  L1_WK0, L1_WK1S, L1_WK1Y, L1_WK2S, L1_WK2Y, L1_WO0, L1_WO1, L1_WO2,
  L1_WQ0, L1_WQ1, L1_WQ2, L1_WR1, L1_WR2, L1_WS0, L1_WS1, L1_WS2,
  L1_WV0, L1_WV1S, L1_WV1Y, L1_WV2S, L1_WV2Y, L1_A1, L1_A2, L1_B1, L1_B2,
  L1_BR1, L1_BR2
};

extern "C" void kernel_launch(void* const* d_in, const int* in_sizes, int n_in,
                              void* d_out, int out_size, void* d_ws, size_t ws_size,
                              hipStream_t stream)
{
  (void)in_sizes; (void)n_in; (void)out_size; (void)ws_size;
  const int E = NN_EDGES, Nn = NN_NODES;
  const float* node_attr = (const float*)d_in[IN_NODE_ATTR];
  const float* edge_attr = (const float*)d_in[IN_EDGE_ATTR];
  const float* pos       = (const float*)d_in[IN_POS];
  const int*   src       = (const int*)d_in[IN_SRC];
  const int*   dst       = (const int*)d_in[IN_DST];
  auto P = [&](int i) { return (const float*)d_in[i]; };

  // Workspace carve-out (floats). ~231 MB total.
  float* ws = (float*)d_ws;
  size_t off = 0;
  auto alloc = [&](size_t n) { float* p = ws + off; off += n; return p; };
  float* Y1  = alloc((size_t)3 * E);
  float* Y2  = alloc((size_t)5 * E);
  float* rr  = alloc((size_t)E);
  float* lg  = alloc((size_t)4 * E);
  float* aa  = alloc((size_t)4 * E);
  float* v0  = alloc((size_t)32 * E);
  float* v1  = alloc((size_t)96 * E);
  float* v2  = alloc((size_t)160 * E);
  float* mx  = alloc((size_t)4 * Nn);
  float* zz  = alloc((size_t)4 * Nn);
  float* o0  = alloc((size_t)32 * Nn);
  float* o1  = alloc((size_t)96 * Nn);
  float* o2  = alloc((size_t)160 * Nn);
  float* q0  = alloc((size_t)32 * Nn);
  float* q1  = alloc((size_t)96 * Nn);
  float* q2  = alloc((size_t)160 * Nn);
  float* f0A = alloc((size_t)32 * Nn);
  float* f1A = alloc((size_t)96 * Nn);
  float* f2A = alloc((size_t)160 * Nn);
  float* f0B = alloc((size_t)32 * Nn);

  const int GM = (Nn + 63) / 64;  // node GEMM grid (4 waves x 16 rows / block)
  const int GE = (E + 63) / 64;   // edge kernel grid
#define FILL(p, v, n) k_fill<<<((n) + 255) / 256, 256, 0, stream>>>((p), (v), (n))

  // -------- geometry --------
  k_geom<<<(E + 255) / 256, 256, 0, stream>>>(pos, src, dst, Y1, Y2, rr, E);

  // -------- layer 0 --------
  k_gemm<<<GM, 128, 0, stream>>>(node_attr, 15, 1, 0, 15, P(L0_WQ0),
                                 nullptr, 0, 0, 0, 0, nullptr, nullptr,
                                 q0, 32, 1, 0, Nn, 32, 0);
  FILL(mx, -1e30f, 4 * Nn);
  FILL(zz, 0.f, 4 * Nn);
  FILL(o0, 0.f, 32 * Nn);
  FILL(o1, 0.f, 96 * Nn);
  FILL(o2, 0.f, 160 * Nn);
  k_l0_edge<<<GE, 128, 0, stream>>>(node_attr, edge_attr, src, dst, q0, Y1, Y2, rr,
                                    P(L0_WR1), P(L0_BR1), P(L0_WR2), P(L0_BR2),
                                    P(L0_WK0), P(L0_WV0), P(L0_WV1), P(L0_WV2),
                                    v0, v1, v2, lg, E);
  k_smax_max<<<(4 * E + 255) / 256, 256, 0, stream>>>(lg, dst, mx, E);
  k_smax_exp<<<(4 * E + 255) / 256, 256, 0, stream>>>(lg, dst, mx, aa, zz, E);
  k_agg<<<(32 * E + 255) / 256, 256, 0, stream>>>(aa, zz, dst, v0, v1, v2,
                                                  o0, o1, o2, E, 1);
  // f0A = relu(o0@Wo0 + node_attr@Ws0)
  k_gemm<<<GM, 128, 0, stream>>>(o0, 32, 1, 0, 32, P(L0_WO0),
                                 node_attr, 15, 1, 0, 15, P(L0_WS0), nullptr,
                                 f0A, 32, 1, 0, Nn, 32, 1);
  for (int m = 0; m < 3; ++m)
    k_gemm<<<GM, 128, 0, stream>>>(o1, 96, 3, m, 32, P(L0_WO1),
                                   nullptr, 0, 0, 0, 0, nullptr, nullptr,
                                   f1A, 96, 3, m, Nn, 32, 0);
  k_gate<<<(32 * Nn + 255) / 256, 256, 0, stream>>>(f1A, P(L0_A1), P(L0_B1), Nn, 3);
  for (int m = 0; m < 5; ++m)
    k_gemm<<<GM, 128, 0, stream>>>(o2, 160, 5, m, 32, P(L0_WO2),
                                   nullptr, 0, 0, 0, 0, nullptr, nullptr,
                                   f2A, 160, 5, m, Nn, 32, 0);
  k_gate<<<(32 * Nn + 255) / 256, 256, 0, stream>>>(f2A, P(L0_A2), P(L0_B2), Nn, 5);

  // -------- layer 1 --------
  k_gemm<<<GM, 128, 0, stream>>>(f0A, 32, 1, 0, 32, P(L1_WQ0),
                                 nullptr, 0, 0, 0, 0, nullptr, nullptr,
                                 q0, 32, 1, 0, Nn, 32, 0);
  for (int m = 0; m < 3; ++m)
    k_gemm<<<GM, 128, 0, stream>>>(f1A, 96, 3, m, 32, P(L1_WQ1),
                                   nullptr, 0, 0, 0, 0, nullptr, nullptr,
                                   q1, 96, 3, m, Nn, 32, 0);
  for (int m = 0; m < 5; ++m)
    k_gemm<<<GM, 128, 0, stream>>>(f2A, 160, 5, m, 32, P(L1_WQ2),
                                   nullptr, 0, 0, 0, 0, nullptr, nullptr,
                                   q2, 160, 5, m, Nn, 32, 0);
  FILL(mx, -1e30f, 4 * Nn);
  FILL(zz, 0.f, 4 * Nn);
  FILL(o0, 0.f, 32 * Nn);
  k_l1_edge<<<GE, 128, 0, stream>>>(f0A, f1A, f2A, edge_attr, src, dst,
                                    q0, q1, q2, Y1, Y2, rr,
                                    P(L1_WR1), P(L1_BR1), P(L1_WR2), P(L1_BR2),
                                    P(L1_WK0), P(L1_WV0), P(L1_WK1S), P(L1_WK1Y),
                                    P(L1_WK2S), P(L1_WK2Y),
                                    v0, lg, E);
  k_smax_max<<<(4 * E + 255) / 256, 256, 0, stream>>>(lg, dst, mx, E);
  k_smax_exp<<<(4 * E + 255) / 256, 256, 0, stream>>>(lg, dst, mx, aa, zz, E);
  k_agg<<<(32 * E + 255) / 256, 256, 0, stream>>>(aa, zz, dst, v0, v1, v2,
                                                  o0, o1, o2, E, 0);
  // f0B = relu(o0@Wo0 + f0A@Ws0); f1n/f2n dead -> skipped
  k_gemm<<<GM, 128, 0, stream>>>(o0, 32, 1, 0, 32, P(L1_WO0),
                                 f0A, 32, 1, 0, 32, P(L1_WS0), nullptr,
                                 f0B, 32, 1, 0, Nn, 32, 1);
  // out = f0B@Wout + bout
  k_gemm<<<GM, 128, 0, stream>>>(f0B, 32, 1, 0, 32, P(P_WOUT),
                                 nullptr, 0, 0, 0, 0, nullptr, P(P_BOUT),
                                 (float*)d_out, 32, 1, 0, Nn, 32, 0);
#undef FILL
}
